// SimilarityTreeLSTM_29154238005588
// MI455X (gfx1250) — matrix operand order repaired and checked by reading, not compiled, not run on hardware
//
#include <hip/hip_runtime.h>

// ---------------------------------------------------------------------------
// SimilarityTreeLSTM on MI455X (gfx1250, wave32).
// Phases:
//  P0 prep    : repack weights (W1 bf16 for WMMA, W2/A1 transposed fp32, A2 bf16)
//  P1 gather  : embedding gather -> 4 bf16 sequences [lemb, remb, lflip, rflip]
//  P2 gemm    : X1[s] = seq[s] @ W1^T + (ioux_b+iouh_b ++ fx_b+fh_b)   (WMMA bf16)
//  P3 chains  : 4 attention-free LSTM chains, one 1024-thr block each
//  P4 combine : hs_l/hs_r = tanh(fwd + rev(bwd)); bf16 copies; row sums
//  P5 gemm    : Pt = (hs @ A2^T + attnh_b)^T                           (WMMA bf16)
//  P6 attn    : 2 full + 2 single-step attention chains (algebraically reduced:
//               h' = h + sum(hmat) - hmat^T softmax(Wa . tanh(P + A1 h)))
//  P7 head    : distance features -> sigmoid MLP -> log_softmax (5 outputs)
// ---------------------------------------------------------------------------

#define L_   512
#define D_   300
#define M_   300
#define G_   1200   // 3M (i,o,u) + M (f)
#define KP_  320    // K padded to multiple of 32 for WMMA
#define H_   200
#define C_   5

typedef __bf16          bf16_t;
typedef bf16_t          v16bf __attribute__((ext_vector_type(16)));
typedef unsigned short  v16us __attribute__((ext_vector_type(16)));
typedef unsigned short  v8us  __attribute__((ext_vector_type(8)));
typedef float           v8f   __attribute__((ext_vector_type(8)));

__device__ __forceinline__ unsigned short f2bf(float f) {
  unsigned int u = __float_as_uint(f);
  return (unsigned short)((u + 0x7FFFu + ((u >> 16) & 1u)) >> 16);  // RNE
}
__device__ __forceinline__ float sigm(float x) { return 1.f / (1.f + __expf(-x)); }

// ---------------------------------------------------------------- P0: prep
__global__ void prep_weights(const float* __restrict__ ioux_w, const float* __restrict__ ioux_b,
                             const float* __restrict__ iouh_w, const float* __restrict__ iouh_b,
                             const float* __restrict__ fx_w,   const float* __restrict__ fx_b,
                             const float* __restrict__ fh_w,   const float* __restrict__ fh_b,
                             const float* __restrict__ attnh_w,const float* __restrict__ attnh_b,
                             unsigned short* __restrict__ W1bf,   // [G_][KP_]
                             float*          __restrict__ W2t,    // [M_][G_] (transposed)
                             float*          __restrict__ A1t,    // [M_][304] (transposed)
                             unsigned short* __restrict__ A2bf,   // [KP_][KP_]
                             float* __restrict__ biasAll,         // [G_]
                             float* __restrict__ attnb)           // [KP_]
{
  int stride = gridDim.x * blockDim.x;
  for (int i = blockIdx.x * blockDim.x + threadIdx.x; i < G_ * KP_; i += stride) {
    { // W1bf: rows 0..899 = ioux_w, 900..1199 = fx_w; K zero-padded 300..319
      int g = i / KP_, k = i % KP_;
      float v = 0.f;
      if (k < D_) v = (g < 3 * M_) ? ioux_w[g * D_ + k] : fx_w[(g - 3 * M_) * D_ + k];
      W1bf[i] = f2bf(v);
    }
    if (i < M_ * G_) { // W2t[k][g]
      int k = i / G_, g = i % G_;
      W2t[i] = (g < 3 * M_) ? iouh_w[g * M_ + k] : fh_w[(g - 3 * M_) * M_ + k];
    }
    if (i < KP_ * KP_) { // A2bf[n][k] = attnh_w[n][M_+k], zero pad
      int n = i / KP_, k = i % KP_;
      float v = (n < M_ && k < M_) ? attnh_w[n * (2 * M_) + M_ + k] : 0.f;
      A2bf[i] = f2bf(v);
    }
    if (i < M_ * 304) { // A1t[k][j] = attnh_w[j][k]
      int k = i / 304, j = i % 304;
      A1t[i] = (j < M_) ? attnh_w[j * (2 * M_) + k] : 0.f;
    }
    if (i < G_) biasAll[i] = (i < 3 * M_) ? (ioux_b[i] + iouh_b[i]) : (fx_b[i - 3 * M_] + fh_b[i - 3 * M_]);
    if (i < KP_) attnb[i] = (i < M_) ? attnh_b[i] : 0.f;
  }
}

// ---------------------------------------------------------------- P1: gather
__global__ void gather_embed(const int* __restrict__ li, const int* __restrict__ ri,
                             const float* __restrict__ emb,
                             unsigned short* __restrict__ seqbf)  // [4][L_][KP_]
{
  int t = blockIdx.x;         // 0..511
  int d = threadIdx.x;        // 0..319
  float lv = 0.f, rv = 0.f;
  if (d < D_) {
    lv = emb[(size_t)li[t] * D_ + d];
    rv = emb[(size_t)ri[t] * D_ + d];
  }
  unsigned short lb = f2bf(lv), rb = f2bf(rv);
  seqbf[((size_t)(0 * L_ + t) * KP_) + d]            = lb;  // lemb
  seqbf[((size_t)(1 * L_ + t) * KP_) + d]            = rb;  // remb
  seqbf[((size_t)(2 * L_ + (L_ - 1 - t)) * KP_) + d] = lb;  // lflip
  seqbf[((size_t)(3 * L_ + (L_ - 1 - t)) * KP_) + d] = rb;  // rflip
}

// ---------------------------------------------------------------- P2/P5: WMMA GEMM
// C[M x N] = A[M x K] @ B[N x K]^T + bias[N];  one wave per 16x16 tile.
// A/B fragments (16-bit A 16x32, B 32x16) use contiguous K: lane half h holds K 16h..16h+15.
__global__ __launch_bounds__(32)
void wmma_gemm_bf16(const unsigned short* __restrict__ Au, int lda,
                    const unsigned short* __restrict__ Bu, int ldb,
                    const float* __restrict__ bias,
                    float* __restrict__ C, int ldc,
                    int ktiles, int storeT)
{
  const int lane = threadIdx.x;
  const int tn = blockIdx.x * 16;
  const int tm = blockIdx.y * 16;
  const bool hi = lane >= 16;
  const int l = lane & 15;

  const unsigned short* arow = Au + (size_t)(tm + l) * lda;
  const unsigned short* brow = Bu + (size_t)(tn + l) * ldb;

  v8f acc = {};
#pragma unroll
  for (int kt = 0; kt < 10; ++kt) {
    if (kt >= ktiles) break;
    const int k0 = kt * 32;
    // A fragment: lo-lanes hold K 0..15, hi-lanes K 16..31
    v8us alo = *(const v8us*)(arow + k0 + (hi ? 16 : 0));
    v8us ahi = *(const v8us*)(arow + k0 + (hi ? 24 : 8));
    v16us au;
#pragma unroll
    for (int i = 0; i < 8; ++i) { au[i] = alo[i]; au[8 + i] = ahi[i]; }
    // B fragment: 16 contiguous K values per lane (lo-lanes K 0..15, hi K 16..31)
    v16us bu = *(const v16us*)(brow + k0 + (hi ? 16 : 0));
    __builtin_prefetch(brow + k0 + 32, 0, 1);  // next k-slab -> global_prefetch

    v16bf a = __builtin_bit_cast(v16bf, au);
    v16bf b = __builtin_bit_cast(v16bf, bu);
    acc = __builtin_amdgcn_wmma_f32_16x16x32_bf16(false, a, false, b,
                                                  (short)0, acc, false, false);
  }

  const int col = tn + l;
  const float bi = bias ? bias[col] : 0.f;
#pragma unroll
  for (int v = 0; v < 8; ++v) {
    const int row = tm + (hi ? v + 8 : v);
    if (storeT) C[(size_t)col * ldc + row] = acc[v] + bi;  // C^T layout
    else        C[(size_t)row * ldc + col] = acc[v] + bi;
  }
}

// ---------------------------------------------------------------- P3: plain chains
__global__ __launch_bounds__(1024)
void chain_fwd(const float* __restrict__ X1,   // [4][L_][G_]
               const float* __restrict__ W2t,  // [M_][G_]
               float* __restrict__ hs)         // [4][L_][M_]
{
  const int s = blockIdx.x;
  const int tid = threadIdx.x;
  __shared__ float h_s[304];
  __shared__ float z_s[1216];
  if (tid < 304) h_s[tid] = 0.f;
  float c = 0.f;
  const float* X  = X1 + (size_t)s * L_ * G_;
  float* out      = hs + (size_t)s * L_ * M_;
  __syncthreads();

  for (int t = 0; t < L_; ++t) {
    const float* xr = X + (size_t)t * G_;
    for (int g = tid; g < G_; g += 1024) {
      float acc = xr[g];
#pragma unroll 4
      for (int k = 0; k < M_; ++k) acc += W2t[k * G_ + g] * h_s[k];  // coalesced
      z_s[g] = acc;
    }
    __syncthreads();
    if (tid < M_) {
      const float i_ = sigm(z_s[tid]);
      const float o_ = sigm(z_s[M_ + tid]);
      const float u_ = tanhf(z_s[2 * M_ + tid]);
      const float f_ = sigm(z_s[3 * M_ + tid]);
      c = i_ * u_ + f_ * c;
      const float h = o_ * tanhf(c);
      h_s[tid] = h;
      out[(size_t)t * M_ + tid] = h;
    }
    __syncthreads();
  }
}

// ---------------------------------------------------------------- P4: combine + sums
__global__ void combine_hs(const float* __restrict__ hs,
                           float* __restrict__ hslr,            // [2][L_][M_]
                           unsigned short* __restrict__ hslrbf) // [2][L_][KP_]
{
  const int t = blockIdx.x, m = threadIdx.x;
  float lv = 0.f, rv = 0.f;
  if (m < M_) {
    lv = tanhf(hs[(size_t)(0 * L_ + t) * M_ + m] + hs[(size_t)(2 * L_ + (L_ - 1 - t)) * M_ + m]);
    rv = tanhf(hs[(size_t)(1 * L_ + t) * M_ + m] + hs[(size_t)(3 * L_ + (L_ - 1 - t)) * M_ + m]);
    hslr[(size_t)(0 * L_ + t) * M_ + m] = lv;
    hslr[(size_t)(1 * L_ + t) * M_ + m] = rv;
  }
  hslrbf[(size_t)(0 * L_ + t) * KP_ + m] = f2bf(lv);
  hslrbf[(size_t)(1 * L_ + t) * KP_ + m] = f2bf(rv);
}

__global__ void rowsum_hs(const float* __restrict__ hslr, float* __restrict__ Hsum) // [2][M_]
{
  const int b = blockIdx.x, m = threadIdx.x;
  if (m < M_) {
    float s = 0.f;
    for (int t = 0; t < L_; ++t) s += hslr[(size_t)(b * L_ + t) * M_ + m];
    Hsum[b * M_ + m] = s;
  }
}

// ---------------------------------------------------------------- P6: attention chains
__global__ __launch_bounds__(1024)
void chain_attn(const float* __restrict__ X1,   // [4][L_][G_]
                const float* __restrict__ W2t,  // [M_][G_]
                const float* __restrict__ A1t,  // [M_][304]
                const float* __restrict__ Wa,   // [M_]
                const float* __restrict__ Pt,   // [2][KP_][L_]   (P transposed)
                const float* __restrict__ Hsum, // [2][M_]
                const float* __restrict__ hslr, // [2][L_][M_]
                float* __restrict__ attnout)    // [4][M_]
{
  const int b = blockIdx.x, tid = threadIdx.x;
  const int sidx_[4] = {1, 0, 3, 2};   // remb, lemb, rflip, lflip
  const int mem_[4]  = {0, 1, 0, 1};   // hs_l, hs_r, hs_l, hs_r
  const int nst_[4]  = {L_, L_, 1, 1}; // rh_b1 / lh_b2 only need step 0
  const float* X  = X1   + (size_t)sidx_[b] * L_ * G_;
  const float* Pm = Pt   + (size_t)mem_[b] * KP_ * L_;
  const float* hm = hslr + (size_t)mem_[b] * L_ * M_;
  const float* Hs = Hsum + mem_[b] * M_;
  float* out = attnout + b * M_;
  const int nsteps = nst_[b];

  __shared__ float h_s[304], hpre_s[304], q_s[304], wa_s[304], hsum_s[304];
  __shared__ float z_s[1216];
  __shared__ float sc_s[512], red_s[512];
  if (tid < 304) {
    h_s[tid]    = 0.f;
    wa_s[tid]   = (tid < M_) ? Wa[tid] : 0.f;
    hsum_s[tid] = (tid < M_) ? Hs[tid] : 0.f;
  }
  float c = 0.f;
  __syncthreads();

  for (int t = 0; t < nsteps; ++t) {
    // --- LSTM gates: z = X1[t] + h @ W2^T
    const float* xr = X + (size_t)t * G_;
    for (int g = tid; g < G_; g += 1024) {
      float acc = xr[g];
#pragma unroll 4
      for (int k = 0; k < M_; ++k) acc += W2t[k * G_ + g] * h_s[k];
      z_s[g] = acc;
    }
    __syncthreads();
    if (tid < M_) {
      const float i_ = sigm(z_s[tid]);
      const float o_ = sigm(z_s[M_ + tid]);
      const float u_ = tanhf(z_s[2 * M_ + tid]);
      const float f_ = sigm(z_s[3 * M_ + tid]);
      c = i_ * u_ + f_ * c;
      hpre_s[tid] = o_ * tanhf(c);
    }
    __syncthreads();
    // --- q = A1 . h_pre  (coalesced via A1t)
    if (tid < M_) {
      float acc = 0.f;
#pragma unroll 4
      for (int k = 0; k < M_; ++k) acc += A1t[k * 304 + tid] * hpre_s[k];
      q_s[tid] = acc;
    }
    __syncthreads();
    // --- scores[j] = Wa . tanh(P[j] + q)   (Pt coalesced over j)
    if (tid < L_) {
      float s = 0.f;
      for (int mm = 0; mm < M_; ++mm)
        s += wa_s[mm] * tanhf(Pm[(size_t)mm * L_ + tid] + q_s[mm]);
      sc_s[tid] = s;
    }
    __syncthreads();
    // --- softmax over 512 (tree reductions)
    if (tid < L_) red_s[tid] = sc_s[tid];
    __syncthreads();
    for (int st = 256; st > 0; st >>= 1) {
      if (tid < st) red_s[tid] = fmaxf(red_s[tid], red_s[tid + st]);
      __syncthreads();
    }
    const float mx = red_s[0];
    __syncthreads();
    if (tid < L_) { const float e = __expf(sc_s[tid] - mx); sc_s[tid] = e; red_s[tid] = e; }
    __syncthreads();
    for (int st = 256; st > 0; st >>= 1) {
      if (tid < st) red_s[tid] += red_s[tid + st];
      __syncthreads();
    }
    const float inv = 1.f / red_s[0];
    __syncthreads();
    // --- h' = h_pre + sum(hmat) - hmat^T s    (since sum(s)==1)
    if (tid < M_) {
      float r = 0.f;
      for (int j = 0; j < L_; ++j) r += sc_s[j] * hm[(size_t)j * M_ + tid]; // coalesced
      const float hn = hpre_s[tid] + hsum_s[tid] - r * inv;
      h_s[tid] = hn;
      if (t == nsteps - 1) out[tid] = hn;
    }
    __syncthreads();
  }
}

// ---------------------------------------------------------------- P7: head
__global__ __launch_bounds__(1024)
void head_kernel(const float* __restrict__ hs, const float* __restrict__ attnout,
                 const float* __restrict__ wh_w, const float* __restrict__ wh_b,
                 const float* __restrict__ wp_w, const float* __restrict__ wp_b,
                 float* __restrict__ out)
{
  __shared__ float vec[4 * M_];
  __shared__ float hid[H_];
  __shared__ float lg[8];
  const int tid = threadIdx.x;
  if (tid < 2 * M_) {
    const int m = tid % M_;
    float lh, rh;
    if (tid < M_) {
      lh = tanhf(hs[(size_t)(0 * L_ + (L_ - 1)) * M_ + m] + attnout[1 * M_ + m]); // lh_f1+lh_f2
      rh = tanhf(attnout[0 * M_ + m] + hs[(size_t)(1 * L_ + (L_ - 1)) * M_ + m]); // rh_f1+rh_f2
    } else {
      lh = tanhf(hs[(size_t)(2 * L_ + 0) * M_ + m] + attnout[3 * M_ + m]);        // lh_b1+lh_b2
      rh = tanhf(attnout[2 * M_ + m] + hs[(size_t)(3 * L_ + 0) * M_ + m]);        // rh_b1+rh_b2
    }
    vec[tid]          = lh * rh;
    vec[2 * M_ + tid] = fabsf(lh - rh);
  }
  __syncthreads();
  if (tid < H_) {
    float a = wh_b[tid];
    const float* w = wh_w + (size_t)tid * (4 * M_);
    for (int k = 0; k < 4 * M_; ++k) a += w[k] * vec[k];
    hid[tid] = sigm(a);
  }
  __syncthreads();
  if (tid < C_) {
    float a = wp_b[tid];
    const float* w = wp_w + (size_t)tid * H_;
    for (int k = 0; k < H_; ++k) a += w[k] * hid[k];
    lg[tid] = a;
  }
  __syncthreads();
  if (tid == 0) {
    float mx = lg[0];
    for (int i = 1; i < C_; ++i) mx = fmaxf(mx, lg[i]);
    float s = 0.f;
    for (int i = 0; i < C_; ++i) s += __expf(lg[i] - mx);
    const float ls = logf(s) + mx;
    for (int i = 0; i < C_; ++i) out[i] = lg[i] - ls;
  }
}

// ---------------------------------------------------------------- launch
extern "C" void kernel_launch(void* const* d_in, const int* in_sizes, int n_in,
                              void* d_out, int out_size, void* d_ws, size_t ws_size,
                              hipStream_t stream)
{
  const int*   linputs = (const int*)d_in[0];
  const int*   rinputs = (const int*)d_in[1];
  const float* emb     = (const float*)d_in[2];
  const float* ioux_w  = (const float*)d_in[3];
  const float* ioux_b  = (const float*)d_in[4];
  const float* iouh_w  = (const float*)d_in[5];
  const float* iouh_b  = (const float*)d_in[6];
  const float* fx_w    = (const float*)d_in[7];
  const float* fx_b    = (const float*)d_in[8];
  const float* fh_w    = (const float*)d_in[9];
  const float* fh_b    = (const float*)d_in[10];
  const float* Wa      = (const float*)d_in[11];
  const float* attnh_w = (const float*)d_in[12];
  const float* attnh_b = (const float*)d_in[13];
  const float* wh_w    = (const float*)d_in[14];
  const float* wh_b    = (const float*)d_in[15];
  const float* wp_w    = (const float*)d_in[16];
  const float* wp_b    = (const float*)d_in[17];

  size_t off = 0;
  auto take = [&](size_t bytes) -> char* {
    char* p = (char*)d_ws + off;
    off += (bytes + 255) & ~(size_t)255;
    return p;
  };
  float*          X1      = (float*)         take((size_t)4 * L_ * G_ * 4);
  float*          hs      = (float*)         take((size_t)4 * L_ * M_ * 4);
  float*          hslr    = (float*)         take((size_t)2 * L_ * M_ * 4);
  float*          Pt      = (float*)         take((size_t)2 * KP_ * L_ * 4);
  float*          Hsum    = (float*)         take((size_t)2 * M_ * 4);
  float*          attnout = (float*)         take((size_t)4 * M_ * 4);
  float*          biasAll = (float*)         take((size_t)G_ * 4);
  float*          attnb   = (float*)         take((size_t)KP_ * 4);
  float*          W2t     = (float*)         take((size_t)M_ * G_ * 4);
  float*          A1t     = (float*)         take((size_t)M_ * 304 * 4);
  unsigned short* seqbf   = (unsigned short*)take((size_t)4 * L_ * KP_ * 2);
  unsigned short* W1bf    = (unsigned short*)take((size_t)G_ * KP_ * 2);
  unsigned short* A2bf    = (unsigned short*)take((size_t)KP_ * KP_ * 2);
  unsigned short* hslrbf  = (unsigned short*)take((size_t)2 * L_ * KP_ * 2);

  prep_weights<<<512, 256, 0, stream>>>(ioux_w, ioux_b, iouh_w, iouh_b, fx_w, fx_b,
                                        fh_w, fh_b, attnh_w, attnh_b,
                                        W1bf, W2t, A1t, A2bf, biasAll, attnb);
  gather_embed<<<L_, KP_, 0, stream>>>(linputs, rinputs, emb, seqbf);

  // X1[s] = seq[s] @ W1^T + biasAll   (M=512, N=1200, K=320)
  for (int s = 0; s < 4; ++s)
    wmma_gemm_bf16<<<dim3(G_ / 16, L_ / 16), 32, 0, stream>>>(
        seqbf + (size_t)s * L_ * KP_, KP_, W1bf, KP_, biasAll,
        X1 + (size_t)s * L_ * G_, G_, KP_ / 32, 0);

  chain_fwd<<<4, 1024, 0, stream>>>(X1, W2t, hs);
  combine_hs<<<L_, KP_, 0, stream>>>(hs, hslr, hslrbf);
  rowsum_hs<<<2, KP_, 0, stream>>>(hslr, Hsum);

  // Pt[m] = (hs_m @ A2^T + attnh_b)^T   (M=512, N=320, K=320, stored transposed)
  for (int m = 0; m < 2; ++m)
    wmma_gemm_bf16<<<dim3(KP_ / 16, L_ / 16), 32, 0, stream>>>(
        hslrbf + (size_t)m * L_ * KP_, KP_, A2bf, KP_, attnb,
        Pt + (size_t)m * KP_ * L_, L_, KP_ / 32, 1);

  chain_attn<<<4, 1024, 0, stream>>>(X1, W2t, A1t, Wa, Pt, Hsum, hslr, attnout);
  head_kernel<<<1, 1024, 0, stream>>>(hs, attnout, wh_w, wh_b, wp_w, wp_b, (float*)d_out);

  (void)in_sizes; (void)n_in; (void)out_size; (void)ws_size;
}
